// SpatialAttentionPool_39127152067250
// MI455X (gfx1250) — compile-verified, matrix operand order checked by reading, not run on hardware
//
#include <hip/hip_runtime.h>
#include <hip/hip_bf16.h>

typedef __attribute__((ext_vector_type(16))) __bf16 v16bf;
typedef __attribute__((ext_vector_type(8)))  float  v8f;

#define BT_GROUPS 128
#define P_DIM 2048
#define C_DIM 256
#define H_DIM 64
#define THREADS 256
#define WAVES 8

__global__ __launch_bounds__(THREADS)
void spatial_attn_pool_kernel(const float* __restrict__ x,
                              const float* __restrict__ W1,
                              const float* __restrict__ b1,
                              const float* __restrict__ W2,
                              float* __restrict__ pooled,
                              float* __restrict__ weights)
{
    // 32KB: W1 as bf16 WMMA B-fragments (32 frags x 32 lanes x 16 elems)
    __shared__ alignas(32) __bf16 ldsW[32 * 512];
    // 16KB: first W1 fp32 async-staging chunks, later scores[2048] + wlds[2048]
    __shared__ alignas(16) float  pool[4096];
    __shared__ float red[16];

    const int tid  = threadIdx.x;
    const int lane = tid & 31;
    const int wave = tid >> 5;
    const int lcol = lane & 15;       // M (A) / N (B) index within half
    const int half = lane >> 4;       // 16-lane half select
    const int kB   = half * 8;        // K sub-base for 16-bit A/B layout
    const int g    = blockIdx.x;

    const float* xg = x + (size_t)g * (size_t)(P_DIM * C_DIM);

    // ---- stage W1 -> LDS via async copy (4 x 16KB chunks), pack to bf16 frags ----
    const unsigned poolBase = (unsigned)(uintptr_t)(&pool[0]);
    for (int cc = 0; cc < 4; ++cc) {
        // async copy W1 rows [cc*64, cc*64+64) : 16KB, 4 x b128 per thread
        #pragma unroll
        for (int r = 0; r < 4; ++r) {
            const unsigned off      = (unsigned)((r * THREADS + tid) * 16);
            const unsigned ldsAddr  = poolBase + off;
            const unsigned long long gAddr =
                (unsigned long long)(uintptr_t)((const char*)(W1 + cc * 4096) + off);
            asm volatile("global_load_async_to_lds_b128 %0, %1, off"
                         :: "v"(ldsAddr), "v"(gAddr) : "memory");
        }
        asm volatile("s_wait_asynccnt 0x0" ::: "memory");
        __syncthreads();

        // pack this chunk into fragments f = cc*8 .. cc*8+7  (kc in {2cc, 2cc+1})
        for (int idx = tid; idx < 4096; idx += THREADS) {
            const int f  = cc * 8 + (idx >> 9);
            const int l  = (idx >> 4) & 31;   // fragment lane
            const int i  = idx & 15;          // element within lane
            const int kc = f >> 2;
            const int n  = f & 3;
            const int kb = (l >> 4) * 8;
            const int kl = (i < 8) ? (kb + i) : (16 + kb + (i - 8));
            const int k  = kc * 32 + kl;      // global K row of W1
            const int h  = n * 16 + (l & 15);
            ldsW[f * 512 + (idx & 511)] = (__bf16)pool[(k - cc * 64) * H_DIM + h];
        }
        __syncthreads();
    }

    // per-lane bias / W2 values for the 4 n-tiles (column = n*16 + lcol)
    float b1v[4], w2v[4];
    #pragma unroll
    for (int n = 0; n < 4; ++n) {
        b1v[n] = b1[n * 16 + lcol];
        w2v[n] = W2[n * 16 + lcol];
    }

    // hoist all 32 B-fragments into registers (loop-invariant across P tiles)
    v16bf bfrag[32];
    #pragma unroll
    for (int f = 0; f < 32; ++f)
        bfrag[f] = *(const v16bf*)&ldsW[f * 512 + lane * 16];

    float* scoresS = pool;          // reuse staging pool (staging is done)
    float* wldsS   = pool + 2048;

    // ---- pass 1: scores = tanh(X@W1 + b1) @ W2 via bf16 WMMA ----
    for (int it = 0; it < 16; ++it) {
        const int pBase  = it * 128 + wave * 16;
        const int row    = pBase + lcol;           // A: M = lane % 16
        const float* xr  = xg + (size_t)row * C_DIM;
        if (it + 1 < 16)
            __builtin_prefetch(xr + 128 * C_DIM, 0, 0);

        v8f z = {};
        v8f acc[4];
        acc[0] = z; acc[1] = z; acc[2] = z; acc[3] = z;

        #pragma unroll
        for (int kc = 0; kc < 8; ++kc) {
            const float* base = xr + kc * 32 + kB;
            float4 f0 = ((const float4*)base)[0];
            float4 f1 = ((const float4*)base)[1];
            float4 f2 = ((const float4*)(base + 16))[0];
            float4 f3 = ((const float4*)(base + 16))[1];
            v16bf a;
            a[0]=(__bf16)f0.x;  a[1]=(__bf16)f0.y;  a[2]=(__bf16)f0.z;  a[3]=(__bf16)f0.w;
            a[4]=(__bf16)f1.x;  a[5]=(__bf16)f1.y;  a[6]=(__bf16)f1.z;  a[7]=(__bf16)f1.w;
            a[8]=(__bf16)f2.x;  a[9]=(__bf16)f2.y;  a[10]=(__bf16)f2.z; a[11]=(__bf16)f2.w;
            a[12]=(__bf16)f3.x; a[13]=(__bf16)f3.y; a[14]=(__bf16)f3.z; a[15]=(__bf16)f3.w;

            #pragma unroll
            for (int n = 0; n < 4; ++n) {
                acc[n] = __builtin_amdgcn_wmma_f32_16x16x32_bf16(
                    false, a, false, bfrag[kc * 4 + n], (short)0, acc[n],
                    false, false);
            }
        }

        // C/D layout: VGPR r of lane L holds (M = half*8 + r, N = L%16)
        #pragma unroll
        for (int r = 0; r < 8; ++r) {
            float s = 0.f;
            #pragma unroll
            for (int n = 0; n < 4; ++n) {
                float hv = acc[n][r] + b1v[n];
                s += tanhf(hv) * w2v[n];
            }
            s += __shfl_xor(s, 1, 32);
            s += __shfl_xor(s, 2, 32);
            s += __shfl_xor(s, 4, 32);
            s += __shfl_xor(s, 8, 32);
            if (lcol == 0)
                scoresS[pBase + half * 8 + r] = s;
        }
    }
    __syncthreads();

    // ---- softmax over P in LDS ----
    float lm = -3.402823466e38f;
    for (int p = tid; p < P_DIM; p += THREADS) lm = fmaxf(lm, scoresS[p]);
    #pragma unroll
    for (int off = 16; off; off >>= 1) lm = fmaxf(lm, __shfl_xor(lm, off, 32));
    if (lane == 0) red[wave] = lm;
    __syncthreads();
    if (tid == 0) {
        float m = red[0];
        #pragma unroll
        for (int i = 1; i < WAVES; ++i) m = fmaxf(m, red[i]);
        red[8] = m;
    }
    __syncthreads();
    const float m = red[8];

    float lsum = 0.f;
    for (int p = tid; p < P_DIM; p += THREADS) lsum += expf(scoresS[p] - m);
    #pragma unroll
    for (int off = 16; off; off >>= 1) lsum += __shfl_xor(lsum, off, 32);
    if (lane == 0) red[wave] = lsum;
    __syncthreads();
    if (tid == 0) {
        float s = 0.f;
        #pragma unroll
        for (int i = 0; i < WAVES; ++i) s += red[i];
        red[9] = s;
    }
    __syncthreads();
    const float inv = 1.0f / red[9];

    // ---- weights out + stage in LDS ----
    float* wOut = weights + (size_t)g * P_DIM;
    for (int p = tid; p < P_DIM; p += THREADS) {
        float w = expf(scoresS[p] - m) * inv;
        wldsS[p] = w;
        wOut[p] = w;
    }
    __syncthreads();

    // ---- pooled[c] = sum_p w[p] * x[p,c]  (one channel per thread, L2-resident) ----
    const int c = tid;
    float accp = 0.f;
    const float* xc = xg + c;
    #pragma unroll 8
    for (int p = 0; p < P_DIM; ++p)
        accp = fmaf(wldsS[p], xc[(size_t)p * C_DIM], accp);
    pooled[(size_t)g * C_DIM + c] = accp;
}

extern "C" void kernel_launch(void* const* d_in, const int* in_sizes, int n_in,
                              void* d_out, int out_size, void* d_ws, size_t ws_size,
                              hipStream_t stream) {
    const float* x  = (const float*)d_in[0];
    const float* W1 = (const float*)d_in[1];
    const float* b1 = (const float*)d_in[2];
    const float* W2 = (const float*)d_in[3];
    // d_in[4] = b2: additive constant on scores -> cancels in softmax; unused.
    float* out     = (float*)d_out;
    float* pooled  = out;                               // [128, 256]
    float* weights = out + (size_t)BT_GROUPS * C_DIM;   // [128, 2048]

    spatial_attn_pool_kernel<<<BT_GROUPS, THREADS, 0, stream>>>(
        x, W1, b1, W2, pooled, weights);
}
